// loss_76055280877598
// MI455X (gfx1250) — matrix operand order repaired and checked by reading, not compiled, not run on hardware
//
#include <hip/hip_runtime.h>

// ---------------------------------------------------------------------------
// CDNA5 (gfx1250) WMMA types
// ---------------------------------------------------------------------------
typedef __attribute__((ext_vector_type(16))) __bf16          v16bf;
typedef __attribute__((ext_vector_type(2)))  __bf16          v2bf;
typedef __attribute__((ext_vector_type(2)))  float           v2f;
typedef __attribute__((ext_vector_type(16))) unsigned short  v16us;
typedef __attribute__((ext_vector_type(8)))  unsigned short  v8us;
typedef __attribute__((ext_vector_type(8)))  float           v8f;

// packed f32x2 -> bf16x2 (RTE); lowers to a single v_cvt_pk_bf16_f32
__device__ __forceinline__ unsigned pack2(float a, float b) {
  v2f vf; vf.x = a; vf.y = b;
  v2bf vb = __builtin_convertvector(vf, v2bf);
  return __builtin_bit_cast(unsigned, vb);
}

__device__ __forceinline__ v8f wmma_bf16(v16us a, v16us b, v8f c) {
  return __builtin_amdgcn_wmma_f32_16x16x32_bf16(
      false, __builtin_bit_cast(v16bf, a),
      false, __builtin_bit_cast(v16bf, b),
      (short)0, c, false, false);
}
__device__ __forceinline__ v16us cat8(v8us lo, v8us hi) {
  return __builtin_shufflevector(lo, hi, 0, 1, 2, 3, 4, 5, 6, 7,
                                 8, 9, 10, 11, 12, 13, 14, 15);
}

// ---------------------------------------------------------------------------
// Implicit-GEMM 3x3 SAME conv + bias + ReLU (NCHW f32 in/out, bf16 WMMA core)
//   M = Cout, N = H*W.  Block tile 64x128, wave tile 32x64.
//   K ordering (main path): k = tap*Cin + cin  -> tap constant per 32-step,
//   all tap geometry hoisted to SALU / once-per-step.  Layer 0 (Cin=3) uses
//   the cin-major ordering in a dedicated tail path (A/B consistent).
//   Requires: Cout%64==0, (H*W)%128==0, W%128==0  (true for every layer).
// ---------------------------------------------------------------------------
__global__ __launch_bounds__(128) void conv3x3_bias_relu_wmma(
    const float* __restrict__ in, const float* __restrict__ wgt,
    const float* __restrict__ bias, float* __restrict__ out,
    int Cin, int Cout, int Hh, int Ww) {
  __shared__ __align__(16) unsigned short As[64][40];    // A[m][k],  80B row
  __shared__ __align__(16) unsigned short Bt[128][40];   // B^T[n][k], 80B row

  const int b   = blockIdx.z;
  const int n0  = blockIdx.x * 128;
  const int m0  = blockIdx.y * 64;
  const int HW  = Hh * Ww;
  const int K   = Cin * 9;
  const bool ktail = (K & 31) != 0;       // only layer 0 (K = 27, Cin = 3)
  const int lc   = (Cin & (Cin - 1)) ? 0 : (31 - __builtin_clz(Cin));
  const int tid  = threadIdx.x;
  const int lane = tid & 31;
  const int wv   = tid >> 5;
  const int wm   = (wv >> 1) * 32;
  const int wn   = (wv & 1) * 64;
  const int ybase = n0 / Ww;              // uniform: W%128==0, tile width 128
  const int x0    = n0 - ybase * Ww;
  const int xn    = x0 - 1 + tid;         // this thread's column x (pre-tap)

  v8f acc[2][4];
  #pragma unroll
  for (int i = 0; i < 2; ++i)
    #pragma unroll
    for (int j = 0; j < 4; ++j)
      #pragma unroll
      for (int e = 0; e < 8; ++e) acc[i][j][e] = 0.f;

  for (int k0 = 0; k0 < K; k0 += 32) {
    if (!ktail) {
      // ---- tap-major: tap r constant for the whole 32-wide K step ----
      const int r  = k0 >> lc;                 // k / Cin   (SALU)
      const int cb = k0 - (r << lc);           // k % Cin
      const int ry = r / 3, rx = r - ry * 3;   // once per step
      const int y  = ybase + ry - 1;
      const int yc = y < 0 ? 0 : (y >= Hh ? Hh - 1 : y);
      const bool oky = (unsigned)y < (unsigned)Hh;
      // A tile: w[m][cin][tap] -> stride-9 scalar loads (weights are tiny)
      #pragma unroll
      for (int it = 0; it < 4; ++it) {
        int i = tid + it * 128;                 // 0..511
        int m = i >> 3, q = (i & 7) << 2;       // k = q..q+3
        int wb = (m0 + m) * K + (cb + q) * 9 + r;
        float e0 = wgt[wb], e1 = wgt[wb + 9], e2 = wgt[wb + 18],
              e3 = wgt[wb + 27];
        unsigned* dst = (unsigned*)&As[m][q];
        dst[0] = pack2(e0, e1);
        dst[1] = pack2(e2, e3);
      }
      // B tile: thread owns column n = tid; x clamp & predicate ONCE per step
      {
        const int xj = xn + rx;                              // per-lane
        const int xc = xj < 0 ? 0 : (xj >= Ww ? Ww - 1 : xj);
        const bool ok = oky & ((unsigned)xj < (unsigned)Ww);
        const int pbase = ((b * Cin + cb) * Hh + yc) * Ww;   // SGPR
        #pragma unroll
        for (int kq = 0; kq < 32; kq += 4) {
          float e[4];
          #pragma unroll
          for (int j = 0; j < 4; ++j)
            e[j] = in[pbase + (kq + j) * HW + xc];  // saddr + fixed voffset
          unsigned p0 = pack2(e[0], e[1]);
          unsigned p1 = pack2(e[2], e[3]);
          unsigned* dst = (unsigned*)&Bt[tid][kq];
          dst[0] = ok ? p0 : 0u;                    // packed select (bf16x2)
          dst[1] = ok ? p1 : 0u;
        }
      }
    } else {
      // ---- layer 0 tail: cin-major k = cin*9 + tap, K=27, one step ----
      #pragma unroll
      for (int it = 0; it < 4; ++it) {
        int i = tid + it * 128;
        int m = i >> 3, q = (i & 7) << 2;
        int wbase = (m0 + m) * K;
        float e[4];
        #pragma unroll
        for (int j = 0; j < 4; ++j) {
          int gk = q + j;
          int ck = gk < K ? gk : K - 1;
          float v = wgt[wbase + ck];
          e[j] = gk < K ? v : 0.f;
        }
        unsigned* dst = (unsigned*)&As[m][q];
        dst[0] = pack2(e[0], e[1]);
        dst[1] = pack2(e[2], e[3]);
      }
      #pragma unroll
      for (int kq = 0; kq < 32; kq += 4) {
        float e[4];
        #pragma unroll
        for (int j = 0; j < 4; ++j) {
          int gk  = kq + j;
          int gkc = gk < K ? gk : K - 1;                // uniform
          int cin = gkc / 9, rr = gkc - cin * 9;        // uniform
          int ry = rr / 3, rx = rr - ry * 3;            // uniform
          int y  = ybase + ry - 1;
          int yc = y < 0 ? 0 : (y >= Hh ? Hh - 1 : y);
          const float* rowp = in + ((b * Cin + cin) * Hh + yc) * Ww;
          bool rowok = (gk < K) & ((unsigned)y < (unsigned)Hh);
          int xj = xn + rx;
          int xc = xj < 0 ? 0 : (xj >= Ww ? Ww - 1 : xj);
          float v = rowp[xc];
          e[j] = (rowok & ((unsigned)xj < (unsigned)Ww)) ? v : 0.f;
        }
        unsigned* dst = (unsigned*)&Bt[tid][kq];
        dst[0] = pack2(e[0], e[1]);
        dst[1] = pack2(e[2], e[3]);
      }
    }
    __syncthreads();

    // ---- fragments (CDNA5 16-bit WMMA layouts), contiguous LDS reads ----
    const int arow = lane & 15;
    const int akb  = (lane >> 4) << 3;         // 0 / 8
    const int bcol = lane & 15;
    const int bkl  = (lane >> 4) << 3;         // 0 / 8
    v16us af[2];
    #pragma unroll
    for (int mi = 0; mi < 2; ++mi) {
      int m = wm + mi * 16 + arow;
      af[mi] = cat8(*(const v8us*)&As[m][akb], *(const v8us*)&As[m][akb + 16]);
    }
    v16us bq[4];
    #pragma unroll
    for (int nf = 0; nf < 4; ++nf) {
      int n = wn + nf * 16 + bcol;
      bq[nf] = cat8(*(const v8us*)&Bt[n][bkl], *(const v8us*)&Bt[n][bkl + 16]);
    }
    #pragma unroll
    for (int mi = 0; mi < 2; ++mi)
      #pragma unroll
      for (int nf = 0; nf < 4; ++nf)
        acc[mi][nf] = wmma_bf16(af[mi], bq[nf], acc[mi][nf]);
    __syncthreads();
  }

  // ---- epilogue: bias + relu, 32-bit offsets (exact tiling, no guards) ----
  const int ncol = lane & 15;
  const int mofs = (lane >> 4) << 3;
  const int mb   = m0 + wm + mofs;
  const int nb   = n0 + wn + ncol;
  const int obase = (b * Cout + mb) * HW + nb;
  #pragma unroll
  for (int mi = 0; mi < 2; ++mi)
    #pragma unroll
    for (int r = 0; r < 8; ++r) {
      float bv = bias[mb + mi * 16 + r];
      int   ob = obase + (mi * 16 + r) * HW;
      #pragma unroll
      for (int nf = 0; nf < 4; ++nf) {
        float v = acc[mi][nf][r] + bv;
        out[ob + nf * 16] = v > 0.f ? v : 0.f;
      }
    }
}

// ---------------------------------------------------------------------------
// Fused per-(b,c) Gram-difference style loss:
//   sum |Kp*(Phi_a^T Phi_a - Phi_g^T Phi_g)| * factor  -> atomicAdd(out)
//   Exact tiling: W%64==0, H%32==0.  Tiles stored [col][k]; staging split so
//   each half-block owns one column range -> fixed voffset, uniform row base.
// ---------------------------------------------------------------------------
__global__ __launch_bounds__(128) void gram_style_wmma(
    const float* __restrict__ a, const float* __restrict__ g,
    int Hh, int Ww, float factor, float* __restrict__ out) {
  __shared__ __align__(16) unsigned short Ta[64][40], Tg[64][40];  // cols w0..
  __shared__ __align__(16) unsigned short Ua[64][40], Ug[64][40];  // cols v0..
  __shared__ float red[128];

  const int plane = blockIdx.z * Hh * Ww;     // < 2^31 for all scales
  const int v0 = blockIdx.x * 64;   // Gram col block
  const int w0 = blockIdx.y * 64;   // Gram row block
  const int tid = threadIdx.x, lane = tid & 31, wv = tid >> 5;
  const int wm = (wv >> 1) * 32, wn = (wv & 1) * 32;

  // staging role: waves 0-1 stage the w0 tiles, waves 2-3 the v0 tiles
  const int c  = tid & 63;
  const int t2 = tid >> 6;                        // uniform per wave
  const int cb = (t2 ? v0 : w0) + c;              // fixed 32-bit voffset
  unsigned short (*TA)[40] = t2 ? Ua : Ta;
  unsigned short (*TG)[40] = t2 ? Ug : Tg;

  v8f accA[2][2], accG[2][2];
  #pragma unroll
  for (int i = 0; i < 2; ++i)
    #pragma unroll
    for (int j = 0; j < 2; ++j)
      #pragma unroll
      for (int e = 0; e < 8; ++e) { accA[i][j][e] = 0.f; accG[i][j][e] = 0.f; }

  for (int k0 = 0; k0 < Hh; k0 += 32) {
    #pragma unroll
    for (int kq = 0; kq < 32; kq += 4) {
      float ea[4], eg[4];
      #pragma unroll
      for (int j = 0; j < 4; ++j) {
        int rofs = plane + (k0 + kq + j) * Ww;    // uniform (SGPR)
        ea[j] = a[rofs + cb];
        eg[j] = g[rofs + cb];
      }
      unsigned* da = (unsigned*)&TA[c][kq];
      unsigned* dg = (unsigned*)&TG[c][kq];
      da[0] = pack2(ea[0], ea[1]); da[1] = pack2(ea[2], ea[3]);
      dg[0] = pack2(eg[0], eg[1]); dg[1] = pack2(eg[2], eg[3]);
    }
    __syncthreads();

    const int arow = lane & 15;
    const int akb  = (lane >> 4) << 3;
    const int bcol = lane & 15;
    const int bkl  = (lane >> 4) << 3;
    v16us afA[2], afG[2], bfA[2], bfG[2];
    #pragma unroll
    for (int mi = 0; mi < 2; ++mi) {
      int m = wm + mi * 16 + arow;
      afA[mi] = cat8(*(const v8us*)&Ta[m][akb], *(const v8us*)&Ta[m][akb + 16]);
      afG[mi] = cat8(*(const v8us*)&Tg[m][akb], *(const v8us*)&Tg[m][akb + 16]);
    }
    #pragma unroll
    for (int ni = 0; ni < 2; ++ni) {
      int n = wn + ni * 16 + bcol;
      bfA[ni] = cat8(*(const v8us*)&Ua[n][bkl], *(const v8us*)&Ua[n][bkl + 16]);
      bfG[ni] = cat8(*(const v8us*)&Ug[n][bkl], *(const v8us*)&Ug[n][bkl + 16]);
    }
    #pragma unroll
    for (int mi = 0; mi < 2; ++mi)
      #pragma unroll
      for (int ni = 0; ni < 2; ++ni) {
        accA[mi][ni] = wmma_bf16(afA[mi], bfA[ni], accA[mi][ni]);
        accG[mi][ni] = wmma_bf16(afG[mi], bfG[ni], accG[mi][ni]);
      }
    __syncthreads();
  }

  float s = 0.f;
  #pragma unroll
  for (int mi = 0; mi < 2; ++mi)
    #pragma unroll
    for (int ni = 0; ni < 2; ++ni)
      #pragma unroll
      for (int r = 0; r < 8; ++r)
        s += fabsf(accA[mi][ni][r] - accG[mi][ni][r]);
  red[tid] = s;
  __syncthreads();
  for (int st = 64; st > 0; st >>= 1) {
    if (tid < st) red[tid] += red[tid + st];
    __syncthreads();
  }
  if (tid == 0) atomicAdd(out, red[0] * factor);
}

// ---------------------------------------------------------------------------
// icomp build + l_hole + l_valid (vectorized x4; HW%4==0; 32-bit offsets)
// ---------------------------------------------------------------------------
__global__ __launch_bounds__(256) void comp_hole_valid(
    const float* __restrict__ igt, const float* __restrict__ iout,
    const float* __restrict__ mask, float* __restrict__ icomp,
    float* __restrict__ out, int Bn, int HW, float invN) {
  __shared__ float red[256];
  const int quads = Bn * 3 * HW / 4;
  const float bw = (float)Bn * invN;
  float acc = 0.f;
  for (int qi = blockIdx.x * blockDim.x + threadIdx.x; qi < quads;
       qi += gridDim.x * blockDim.x) {
    int idx = qi * 4;
    int p = idx % HW;
    int b = (idx / HW) / 3;
    int mo = b * HW + p;
    #pragma unroll
    for (int j = 0; j < 4; ++j) {
      float m  = mask[mo + j];
      float gv = igt[idx + j], ov = iout[idx + j];
      bool hole = (m == 0.f);
      icomp[idx + j] = hole ? ov : gv;
      float d = fabsf(ov - gv);
      acc += hole ? d * bw : d * invN;
    }
  }
  red[threadIdx.x] = acc;
  __syncthreads();
  for (int st = 128; st > 0; st >>= 1) {
    if (threadIdx.x < st) red[threadIdx.x] += red[threadIdx.x + st];
    __syncthreads();
  }
  if (threadIdx.x == 0) atomicAdd(out, red[0]);
}

// ---------------------------------------------------------------------------
// 2x2 max-pool, stride 2 (NCHW; 32-bit offsets)
// ---------------------------------------------------------------------------
__global__ __launch_bounds__(256) void maxpool2x2(
    const float* __restrict__ in, float* __restrict__ out,
    int total, int Ho, int Wo) {
  for (int idx = blockIdx.x * blockDim.x + threadIdx.x; idx < total;
       idx += gridDim.x * blockDim.x) {
    int x  = idx % Wo;
    int t  = idx / Wo;
    int y  = t % Ho;
    int bc = t / Ho;
    int base = (bc * 2 * Ho + 2 * y) * 2 * Wo + 2 * x;
    float v0 = in[base], v1 = in[base + 1];
    float v2 = in[base + 2 * Wo], v3 = in[base + 2 * Wo + 1];
    out[idx] = fmaxf(fmaxf(v0, v1), fmaxf(v2, v3));
  }
}

// ---------------------------------------------------------------------------
// scale * sum |a-b| (vectorized x4; n%4==0)  -> atomicAdd(out)
// ---------------------------------------------------------------------------
__global__ __launch_bounds__(256) void l1_scaled(
    const float* __restrict__ a, const float* __restrict__ b,
    int n, float scale, float* __restrict__ out) {
  __shared__ float red[256];
  const int quads = n / 4;
  float acc = 0.f;
  for (int qi = blockIdx.x * blockDim.x + threadIdx.x; qi < quads;
       qi += gridDim.x * blockDim.x) {
    int i = qi * 4;
    acc += fabsf(a[i] - b[i]) + fabsf(a[i + 1] - b[i + 1]) +
           fabsf(a[i + 2] - b[i + 2]) + fabsf(a[i + 3] - b[i + 3]);
  }
  red[threadIdx.x] = acc;
  __syncthreads();
  for (int st = 128; st > 0; st >>= 1) {
    if (threadIdx.x < st) red[threadIdx.x] += red[threadIdx.x + st];
    __syncthreads();
  }
  if (threadIdx.x == 0) atomicAdd(out, red[0] * scale);
}

// ---------------------------------------------------------------------------
// Orchestration
// ---------------------------------------------------------------------------
extern "C" void kernel_launch(void* const* d_in, const int* in_sizes, int n_in,
                              void* d_out, int out_size, void* d_ws, size_t ws_size,
                              hipStream_t stream) {
  (void)in_sizes; (void)n_in; (void)ws_size;
  const int Bn = 2, Him = 512;
  const int HWim = Him * Him;

  const float* igt  = (const float*)d_in[0];
  const float* iout = (const float*)d_in[1];
  const float* mask = (const float*)d_in[2];
  const float* wgt[7];
  const float* bia[7];
  for (int i = 0; i < 7; ++i) {
    wgt[i] = (const float*)d_in[3 + 2 * i];
    bia[i] = (const float*)d_in[4 + 2 * i];
  }
  float* outf = (float*)d_out;
  hipMemsetAsync(d_out, 0, (size_t)out_size * sizeof(float), stream);

  // workspace carve (floats): ~451 MB total
  float* wsf = (float*)d_ws;
  size_t off = 0;
  auto carve = [&](size_t n) { float* p = wsf + off; off += n; return p; };
  const size_t S_IMG = (size_t)Bn * 3 * HWim;
  const size_t S_PP  = (size_t)Bn * 64 * 512 * 512;
  const size_t SP1   = (size_t)Bn * 64 * 256 * 256;
  const size_t SP2   = (size_t)Bn * 128 * 128 * 128;
  const size_t SP3   = (size_t)Bn * 256 * 64 * 64;
  float* icomp = carve(S_IMG);
  float* pa = carve(S_PP);
  float* pb = carve(S_PP);
  float* P[3][3];   // [image: 0=out,1=gt,2=comp][scale]
  for (int im = 0; im < 3; ++im) {
    P[im][0] = carve(SP1); P[im][1] = carve(SP2); P[im][2] = carve(SP3);
  }

  // icomp + l_hole + l_valid
  {
    int total = Bn * 3 * HWim;
    int blocks = (total / 4 + 255) / 256; if (blocks > 2048) blocks = 2048;
    comp_hole_valid<<<blocks, 256, 0, stream>>>(igt, iout, mask, icomp, outf,
                                                Bn, HWim, 1.f / (float)total);
  }

  auto conv = [&](const float* in, int Cin, int Cout, int S,
                  const float* wg, const float* bb, float* o) {
    dim3 grid((S * S) / 128, Cout / 64, Bn);
    conv3x3_bias_relu_wmma<<<grid, 128, 0, stream>>>(in, wg, bb, o, Cin, Cout, S, S);
  };
  auto pool = [&](const float* in, float* o, int C, int Si) {
    int So = Si / 2;
    int total = Bn * C * So * So;
    int blocks = (total + 255) / 256; if (blocks > 4096) blocks = 4096;
    maxpool2x2<<<blocks, 256, 0, stream>>>(in, o, total, So, So);
  };
  auto vgg = [&](const float* img, float** p) {
    conv(img,  3,   64, 512, wgt[0], bia[0], pa);
    conv(pa,   64,  64, 512, wgt[1], bia[1], pb);
    pool(pb, p[0], 64, 512);
    conv(p[0], 64, 128, 256, wgt[2], bia[2], pa);
    conv(pa,  128, 128, 256, wgt[3], bia[3], pb);
    pool(pb, p[1], 128, 256);
    conv(p[1],128, 256, 128, wgt[4], bia[4], pa);
    conv(pa,  256, 256, 128, wgt[5], bia[5], pb);
    conv(pb,  256, 256, 128, wgt[6], bia[6], pa);
    pool(pa, p[2], 256, 128);
  };
  vgg(iout,  P[0]);
  vgg(igt,   P[1]);
  vgg(icomp, P[2]);

  // l_perc: always normalized by prod(pool_gt[0].shape) (faithful to reference)
  const float invNigt = 1.f / (float)SP1;
  const size_t psz[3] = {SP1, SP2, SP3};
  for (int s = 0; s < 3; ++s) {
    int n = (int)psz[s];
    int blocks = (n / 4 + 255) / 256; if (blocks > 4096) blocks = 4096;
    l1_scaled<<<blocks, 256, 0, stream>>>(P[0][s], P[1][s], n, invNigt, outf);
    l1_scaled<<<blocks, 256, 0, stream>>>(P[2][s], P[1][s], n, invNigt, outf);
  }

  // style: sum |Kp*(Ga-Gg)| / C^2 for (comp,gt) and (out,gt) at each scale
  const int Cs[3] = {64, 128, 256};
  const int Ws[3] = {256, 128, 64};
  for (int s = 0; s < 3; ++s) {
    float Kp = 1.f / ((float)Cs[s] * (float)Ws[s] * (float)Ws[s]);
    float factor = Kp / ((float)Cs[s] * (float)Cs[s]);
    dim3 grid(Ws[s] / 64, Ws[s] / 64, Bn * Cs[s]);
    gram_style_wmma<<<grid, 128, 0, stream>>>(P[2][s], P[1][s], Ws[s], Ws[s], factor, outf);
    gram_style_wmma<<<grid, 128, 0, stream>>>(P[0][s], P[1][s], Ws[s], Ws[s], factor, outf);
  }
}
